// SlotAttention_83184926589125
// MI455X (gfx1250) — compile-verified, matrix operand order checked by reading, not run on hardware
//
#include <hip/hip_runtime.h>
#include <hip/hip_bf16.h>

// ---------------------------------------------------------------------------
// SlotAttention fused implementation for gfx1250 (MI455X, wave32, WMMA).
// Heavy GEMMs (feats@Wk, feats@Wv) run as bf16 WMMA (v_wmma_f32_16x16x32_bf16)
// fused with the slot-softmax attention so `features` (64 MB) is read once.
// ---------------------------------------------------------------------------

#define N_SLOTS   8
#define SLOT_DIM  256
#define FEAT_DIM  128
#define N_HEADS   4
#define HEAD_DIM  64
#define BATCH     32
#define HW        4096          // 64*64
#define TILE_N    64            // pixels per workgroup tile
#define N_TILES   (HW / TILE_N) // 64 tiles per batch

typedef __attribute__((ext_vector_type(16))) __bf16 v16bf;
typedef __attribute__((ext_vector_type(8)))  float  v8f;

union Frag32B { uint4 q[2]; v16bf v; };

static __device__ __forceinline__ unsigned short f2bf(float f) {
    unsigned int u = __float_as_uint(f);
    u = (u + 0x7FFFu + ((u >> 16) & 1u)) >> 16;   // round-to-nearest-even
    return (unsigned short)u;
}
static __device__ __forceinline__ float bf2f(unsigned short h) {
    return __uint_as_float(((unsigned int)h) << 16);
}

static __device__ __forceinline__ float blk_sum256(float v, float* red) {
    int d = threadIdx.x;
    red[d] = v; __syncthreads();
    for (int s = 128; s > 0; s >>= 1) {
        if (d < s) red[d] += red[d + s];
        __syncthreads();
    }
    float r = red[0]; __syncthreads();
    return r;
}

// ---------------------------------------------------------------------------
// Kernel 0: pack Wk/Wv (fp32, row-major [K=128][N=256]) into bf16 WMMA
// B-fragment order: frag id = mat*64 + kc*16 + nb ; each frag = 32 lanes x 16
// bf16 halves (contiguous per lane). Lane n<16 holds N=nb*16+n; lane-half
// selects K subranges per ISA 7.12.2 (K = kc*32 + hi*8 + (i<8 ? i : i+8)).
// ---------------------------------------------------------------------------
__global__ void pack_w_kernel(const float* __restrict__ Wk,
                              const float* __restrict__ Wv,
                              unsigned short* __restrict__ wfrag) {
    int g    = blockIdx.x * 256 + threadIdx.x;   // 0 .. 65535
    int mat  = g >> 15;
    int rem  = g & 32767;
    int frag = rem >> 9;            // 0..63
    int lane = (rem >> 4) & 31;
    int i    = rem & 15;
    int kc   = frag >> 4;
    int nb   = frag & 15;
    int hi   = lane >> 4;
    int K    = kc * 32 + hi * 8 + (i < 8 ? i : i + 8);
    int N    = nb * 16 + (lane & 15);
    const float* W = mat ? Wv : Wk;
    wfrag[g] = f2bf(W[(size_t)K * SLOT_DIM + N]);
}

// ---------------------------------------------------------------------------
// Kernel 1: slots_n = LN(slots); q = slots_n @ Wq + bq ; zero agg accumulator.
// One block per (b, slot) row, 256 threads (one per dim).
// ---------------------------------------------------------------------------
__global__ void qprep_kernel(const float* __restrict__ slots,
                             const float* __restrict__ Wq,
                             const float* __restrict__ bq,
                             const float* __restrict__ g_pre,
                             const float* __restrict__ b_pre,
                             float* __restrict__ q_ws,
                             float* __restrict__ agg_ws) {
    __shared__ float sx[256];
    __shared__ float red[256];
    int row = blockIdx.x;           // b*8 + s
    int d   = threadIdx.x;
    float x  = slots[(size_t)row * SLOT_DIM + d];
    float mu = blk_sum256(x, red) * (1.0f / 256.0f);
    float dv = x - mu;
    float var = blk_sum256(dv * dv, red) * (1.0f / 256.0f);
    float sn  = dv * rsqrtf(var + 1e-5f) * g_pre[d] + b_pre[d];
    sx[d] = sn; __syncthreads();
    float acc = 0.0f;
    for (int c = 0; c < 256; ++c) acc += sx[c] * Wq[(size_t)c * SLOT_DIM + d];
    q_ws[(size_t)row * SLOT_DIM + d]   = acc + bq[d];
    agg_ws[(size_t)row * SLOT_DIM + d] = 0.0f;   // zero accumulator for kernel 2
}

// ---------------------------------------------------------------------------
// Kernel 2: fused K/V projection + slot attention over a 64-pixel tile.
// ---------------------------------------------------------------------------
#define A_STRIDE  136   // halves per A row  (128 + 8 pad, 16B aligned rows)
#define KV_STRIDE 264   // halves per KV row (256 + 8 pad)

// GEMM structured to bound VGPR pressure (no spills):
//   nbi loop     : unroll 1 (serialize the two N-blocks)
//   kc loop outer: unroll 1, ONE B-fragment live at a time
//   m loop inner : full unroll, 4 persistent accumulators (32 VGPRs)
static __device__ __forceinline__ void gemm_tile_bf16(
        int mat, const float* __restrict__ bias,
        const unsigned short* __restrict__ lA,
        unsigned short* __restrict__ lKV,
        const unsigned short* __restrict__ wfrag,
        int lane, int w) {
    const int mrow  = lane & 15;
    const int hi8   = (lane >> 4) * 8;
    const int mbase = (lane >> 4) * 8;
#pragma unroll 1
    for (int nbi = 0; nbi < 2; ++nbi) {
        int nb = w * 2 + nbi;
        v8f acc[4];
#pragma unroll
        for (int m = 0; m < 4; ++m)
            acc[m] = (v8f){0.f, 0.f, 0.f, 0.f, 0.f, 0.f, 0.f, 0.f};
#pragma unroll 1
        for (int kc = 0; kc < 4; ++kc) {
            Frag32B bfr;
            const unsigned short* p =
                wfrag + ((size_t)(mat * 64 + kc * 16 + nb)) * 512 + lane * 16;
            bfr.q[0] = *(const uint4*)(p);       // halves 0..7
            bfr.q[1] = *(const uint4*)(p + 8);   // halves 8..15
            const unsigned short* abase = lA + (size_t)mrow * A_STRIDE + kc * 32 + hi8;
#pragma unroll
            for (int m = 0; m < 4; ++m) {
                Frag32B af;
                const unsigned short* ap = abase + (size_t)(m * 16) * A_STRIDE;
                af.q[0] = *(const uint4*)(ap);        // K = kc*32 + hi*8 + 0..7
                af.q[1] = *(const uint4*)(ap + 16);   // K = kc*32 + hi*8 + 16..23
                acc[m] = __builtin_amdgcn_wmma_f32_16x16x32_bf16(
                        false, af.v, false, bfr.v, (short)0, acc[m], false, false);
            }
        }
        int   ncol = nb * 16 + (lane & 15);
        float bb   = bias[ncol];
#pragma unroll
        for (int m = 0; m < 4; ++m) {
#pragma unroll
            for (int r = 0; r < 8; ++r) {
                int rowo = m * 16 + mbase + r;        // C layout: M = r + 8*(lane>=16)
                lKV[(size_t)rowo * KV_STRIDE + ncol] = f2bf(acc[m][r] + bb);
            }
        }
    }
}

__global__ void __launch_bounds__(256)
fused_attn_kernel(const float* __restrict__ features,
                  const float* __restrict__ bk,
                  const float* __restrict__ bv,
                  const unsigned short* __restrict__ wfrag,
                  const float* __restrict__ q_ws,
                  float* __restrict__ agg_ws) {
    __shared__ __align__(16) unsigned short lA[TILE_N * A_STRIDE];    // 17.0 KB
    __shared__ __align__(16) unsigned short lKV[TILE_N * KV_STRIDE];  // 33.0 KB
    __shared__ unsigned short lQ[N_SLOTS * SLOT_DIM];                 //  4.0 KB
    __shared__ float lAttn[N_HEADS * N_SLOTS * TILE_N];               //  8.0 KB

    int blk  = blockIdx.x;
    int b    = blk >> 6;          // /64 tiles
    int tile = blk & 63;
    int n0   = tile * TILE_N;
    int t    = threadIdx.x;
    int lane = t & 31;
    int w    = t >> 5;            // 8 waves

    // ---- stage feature tile (transpose CHW -> [pixel][chan], bf16) ----
    for (int idx = t; idx < TILE_N * FEAT_DIM; idx += 256) {
        int c = idx >> 6;         // channel 0..127
        int n = idx & 63;         // local pixel
        float f = features[((size_t)b * FEAT_DIM + c) * HW + n0 + n];
        lA[(size_t)n * A_STRIDE + c] = f2bf(f);
    }
    // ---- stage q (bf16) ----
    for (int idx = t; idx < N_SLOTS * SLOT_DIM; idx += 256) {
        lQ[idx] = f2bf(q_ws[(size_t)b * N_SLOTS * SLOT_DIM + idx]);
    }
    __syncthreads();

    // ---- K = feats @ Wk + bk  (WMMA) ----
    gemm_tile_bf16(0, bk, lA, lKV, wfrag, lane, w);
    __syncthreads();

    // ---- scores + softmax over the 8 slots (per head, per pixel) ----
    {
        int h = t >> 6;           // head 0..3
        int n = t & 63;           // pixel
        float sc[N_SLOTS];
        float mx = -1e30f;
        const unsigned short* kp = &lKV[(size_t)n * KV_STRIDE + h * HEAD_DIM];
        for (int s = 0; s < N_SLOTS; ++s) {
            const unsigned short* qp = &lQ[s * SLOT_DIM + h * HEAD_DIM];
            float acc = 0.0f;
            for (int d = 0; d < HEAD_DIM; ++d) acc += bf2f(qp[d]) * bf2f(kp[d]);
            sc[s] = acc * 0.125f;                 // HEAD_DIM^-0.5
            mx = fmaxf(mx, sc[s]);
        }
        float sum = 0.0f;
        for (int s = 0; s < N_SLOTS; ++s) { sc[s] = __expf(sc[s] - mx); sum += sc[s]; }
        float inv = 1.0f / sum;
        for (int s = 0; s < N_SLOTS; ++s)
            lAttn[(h * N_SLOTS + s) * TILE_N + n] = sc[s] * inv;
    }
    __syncthreads();

    // ---- V = feats @ Wv + bv  (WMMA, reuses lKV) ----
    gemm_tile_bf16(1, bv, lA, lKV, wfrag, lane, w);
    __syncthreads();

    // ---- agg[s][o] += sum_n attn[h(o)][s][n] * V[n][o] ----
    {
        int o = t;                // 0..255
        int h = o >> 6;
        for (int s = 0; s < N_SLOTS; ++s) {
            const float* ap = &lAttn[(h * N_SLOTS + s) * TILE_N];
            float acc = 0.0f;
            for (int n = 0; n < TILE_N; ++n)
                acc += ap[n] * bf2f(lKV[(size_t)n * KV_STRIDE + o]);
            atomicAdd(&agg_ws[((size_t)b * N_SLOTS + s) * SLOT_DIM + o], acc);
        }
    }
}

// ---------------------------------------------------------------------------
// Kernel 3: gi = agg @ W_ih^T + b_ih ; gh = slots @ W_hh^T + b_hh
// Tiled fp32 GEMM: 192 blocks = 2 mats x 8 row-blocks(32) x 12 col-blocks(64).
// ---------------------------------------------------------------------------
__global__ void gates_gemm_kernel(const float* __restrict__ Xagg,
                                  const float* __restrict__ Xslots,
                                  const float* __restrict__ Wih,
                                  const float* __restrict__ bih,
                                  const float* __restrict__ Whh,
                                  const float* __restrict__ bhh,
                                  float* __restrict__ gi,
                                  float* __restrict__ gh) {
    __shared__ float Xl[32][256];
    __shared__ float Wl[64][65];
    int blk = blockIdx.x;
    int mat = blk / 96;
    int rem = blk % 96;
    int rb  = rem / 12, cb = rem % 12;
    const float* X    = mat ? Xslots : Xagg;
    const float* W    = mat ? Whh    : Wih;
    const float* bias = mat ? bhh    : bih;
    float*       out  = mat ? gh     : gi;
    int r0 = rb * 32, j0 = cb * 64;
    int t  = threadIdx.x;

    for (int idx = t; idx < 32 * 256; idx += 256)
        Xl[idx >> 8][idx & 255] = X[(size_t)(r0 + (idx >> 8)) * 256 + (idx & 255)];

    int col = t & 63, rloc = t >> 6;
    float acc[8];
#pragma unroll
    for (int r = 0; r < 8; ++r) acc[r] = 0.0f;

    for (int c0 = 0; c0 < 256; c0 += 64) {
        __syncthreads();
        for (int idx = t; idx < 64 * 64; idx += 256)
            Wl[idx >> 6][idx & 63] = W[(size_t)(j0 + (idx >> 6)) * 256 + c0 + (idx & 63)];
        __syncthreads();
#pragma unroll
        for (int rr = 0; rr < 8; ++rr) {
            int row = rr * 4 + rloc;
            float a = 0.0f;
            for (int cc = 0; cc < 64; ++cc) a += Xl[row][c0 + cc] * Wl[col][cc];
            acc[rr] += a;
        }
    }
    float bb = bias[j0 + col];
#pragma unroll
    for (int rr = 0; rr < 8; ++rr)
        out[(size_t)(r0 + rr * 4 + rloc) * 768 + j0 + col] = acc[rr] + bb;
}

// ---------------------------------------------------------------------------
// Kernel 4: GRU gates -> updated ; post-LayerNorm -> ln_out.
// ---------------------------------------------------------------------------
__global__ void gru_ln_kernel(const float* __restrict__ gi,
                              const float* __restrict__ gh,
                              const float* __restrict__ slots,
                              const float* __restrict__ g_post,
                              const float* __restrict__ b_post,
                              float* __restrict__ upd,
                              float* __restrict__ lnout) {
    __shared__ float red[256];
    int row = blockIdx.x;
    int d   = threadIdx.x;
    const float* gir = gi + (size_t)row * 768;
    const float* ghr = gh + (size_t)row * 768;
    float i_r = gir[d], i_z = gir[256 + d], i_n = gir[512 + d];
    float h_r = ghr[d], h_z = ghr[256 + d], h_n = ghr[512 + d];
    float r  = 1.0f / (1.0f + __expf(-(i_r + h_r)));
    float z  = 1.0f / (1.0f + __expf(-(i_z + h_z)));
    float nn = tanhf(i_n + r * h_n);
    float hp = slots[(size_t)row * 256 + d];
    float hnew = (1.0f - z) * nn + z * hp;
    upd[(size_t)row * 256 + d] = hnew;

    float mu  = blk_sum256(hnew, red) * (1.0f / 256.0f);
    float dv  = hnew - mu;
    float var = blk_sum256(dv * dv, red) * (1.0f / 256.0f);
    lnout[(size_t)row * 256 + d] = dv * rsqrtf(var + 1e-5f) * g_post[d] + b_post[d];
}

// ---------------------------------------------------------------------------
// Kernel 5: hidden = relu(ln @ W1 + b1)   (256x256 @ 256x512)
// ---------------------------------------------------------------------------
__global__ void mlp1_kernel(const float* __restrict__ ln,
                            const float* __restrict__ W1,
                            const float* __restrict__ b1,
                            float* __restrict__ hid) {
    __shared__ float Xl[32][257];
    int blk = blockIdx.x;            // 64 = 8 rb x 8 cb
    int rb = blk >> 3, cb = blk & 7;
    int r0 = rb * 32, j0 = cb * 64;
    int t  = threadIdx.x;
    for (int idx = t; idx < 32 * 256; idx += 256)
        Xl[idx >> 8][idx & 255] = ln[(size_t)(r0 + (idx >> 8)) * 256 + (idx & 255)];
    __syncthreads();
    int col = t & 63, rloc = t >> 6;
    int j = j0 + col;
    float acc[8];
#pragma unroll
    for (int r = 0; r < 8; ++r) acc[r] = 0.0f;
    for (int c = 0; c < 256; ++c) {
        float wv = W1[(size_t)c * 512 + j];
#pragma unroll
        for (int rr = 0; rr < 8; ++rr) acc[rr] += Xl[rr * 4 + rloc][c] * wv;
    }
    float bb = b1[j];
#pragma unroll
    for (int rr = 0; rr < 8; ++rr)
        hid[(size_t)(r0 + rr * 4 + rloc) * 512 + j] = fmaxf(acc[rr] + bb, 0.0f);
}

// ---------------------------------------------------------------------------
// Kernel 6: out = updated + (hidden @ W2 + b2)   (256x512 @ 512x256)
// ---------------------------------------------------------------------------
__global__ void mlp2_kernel(const float* __restrict__ hid,
                            const float* __restrict__ W2,
                            const float* __restrict__ b2,
                            const float* __restrict__ upd,
                            float* __restrict__ out) {
    __shared__ float Xl[16][513];
    int blk = blockIdx.x;            // 64 = 16 rb x 4 cb
    int rb = blk >> 2, cb = blk & 3;
    int r0 = rb * 16, j0 = cb * 64;
    int t  = threadIdx.x;
    for (int idx = t; idx < 16 * 512; idx += 256)
        Xl[idx >> 9][idx & 511] = hid[(size_t)(r0 + (idx >> 9)) * 512 + (idx & 511)];
    __syncthreads();
    int col = t & 63, rloc = t >> 6;
    int j = j0 + col;
    float acc[4];
#pragma unroll
    for (int r = 0; r < 4; ++r) acc[r] = 0.0f;
    for (int c = 0; c < 512; ++c) {
        float wv = W2[(size_t)c * 256 + j];
#pragma unroll
        for (int rr = 0; rr < 4; ++rr) acc[rr] += Xl[rr * 4 + rloc][c] * wv;
    }
    float bb = b2[j];
#pragma unroll
    for (int rr = 0; rr < 4; ++rr) {
        int row = r0 + rr * 4 + rloc;
        out[(size_t)row * 256 + j] = upd[(size_t)row * 256 + j] + acc[rr] + bb;
    }
}

// ---------------------------------------------------------------------------
extern "C" void kernel_launch(void* const* d_in, const int* in_sizes, int n_in,
                              void* d_out, int out_size, void* d_ws, size_t ws_size,
                              hipStream_t stream) {
    const float* features = (const float*)d_in[0];
    const float* slots    = (const float*)d_in[1];
    const float* Wq       = (const float*)d_in[2];
    const float* bq       = (const float*)d_in[3];
    const float* Wk       = (const float*)d_in[4];
    const float* bk       = (const float*)d_in[5];
    const float* Wv       = (const float*)d_in[6];
    const float* bv       = (const float*)d_in[7];
    const float* Wih      = (const float*)d_in[8];
    const float* bih      = (const float*)d_in[9];
    const float* Whh      = (const float*)d_in[10];
    const float* bhh      = (const float*)d_in[11];
    const float* g_pre    = (const float*)d_in[12];
    const float* b_pre    = (const float*)d_in[13];
    const float* g_post   = (const float*)d_in[14];
    const float* b_post   = (const float*)d_in[15];
    const float* W1       = (const float*)d_in[16];
    const float* b1       = (const float*)d_in[17];
    const float* W2       = (const float*)d_in[18];
    const float* b2       = (const float*)d_in[19];
    float* out = (float*)d_out;

    // workspace layout (~3.2 MB)
    char* ws = (char*)d_ws;
    float* q_ws   = (float*)ws;  ws += (size_t)65536  * 4;  // B*8*256
    float* agg_ws = (float*)ws;  ws += (size_t)65536  * 4;
    float* gi_ws  = (float*)ws;  ws += (size_t)196608 * 4;  // 256*768
    float* gh_ws  = (float*)ws;  ws += (size_t)196608 * 4;
    float* upd_ws = (float*)ws;  ws += (size_t)65536  * 4;
    float* ln_ws  = (float*)ws;  ws += (size_t)65536  * 4;
    float* hid_ws = (float*)ws;  ws += (size_t)131072 * 4;  // 256*512
    unsigned short* wfrag = (unsigned short*)ws;            // 2*64*512 halves

    pack_w_kernel   <<<256,  256, 0, stream>>>(Wk, Wv, wfrag);
    qprep_kernel    <<<256,  256, 0, stream>>>(slots, Wq, bq, g_pre, b_pre, q_ws, agg_ws);
    fused_attn_kernel<<<BATCH * N_TILES, 256, 0, stream>>>(features, bk, bv, wfrag, q_ws, agg_ws);
    gates_gemm_kernel<<<192, 256, 0, stream>>>(agg_ws, slots, Wih, bih, Whh, bhh, gi_ws, gh_ws);
    gru_ln_kernel   <<<256,  256, 0, stream>>>(gi_ws, gh_ws, slots, g_post, b_post, upd_ws, ln_ws);
    mlp1_kernel     <<<64,   256, 0, stream>>>(ln_ws, W1, b1, hid_ws);
    mlp2_kernel     <<<64,   256, 0, stream>>>(hid_ws, W2, b2, upd_ws, out);
}
